// SpeciesMoE_27745488732220
// MI455X (gfx1250) — matrix operand order repaired and checked by reading, not compiled, not run on hardware
//
#include <hip/hip_runtime.h>
#include <hip/hip_bf16.h>

// ---------------------------------------------------------------------------
// MoE forward for gfx1250 (MI455X): bf16 WMMA expert FFN, sparse dispatch,
// M=32 token tiles with shared B-fragments, TDM gather load of the A-tile
// with TDM LDS padding for bank-conflict-free ds_load_b128 fragment reads.
// ---------------------------------------------------------------------------

#define B_  4
#define L_  1024
#define N_  4096        // B*L tokens
#define D_  768
#define H_  1536        // 2*D
#define E_  8
#define K_  2
#define MT_ 32          // tokens per block tile
#define TILES_ (N_ / MT_)   // 128 : max tiles per expert

// LDS layouts (element = bf16):
//  sA row stride: 768 data + 8-elem pad every 256 elems (TDM pad) = 792 elems
//    -> 396 dwords/row, 396 % 64 = 12 banks: lanes 0..15 land on 12*L mod 64,
//       all distinct -> conflict-free b128 reads.
#define AROW_ 792
//  sH row stride: 1536 + 8 = 1544 elems -> 772 dwords, 772 % 64 = 4 banks.
#define HROW_ 1544

typedef __attribute__((ext_vector_type(16))) __bf16   v16bf;
typedef __attribute__((ext_vector_type(8)))  __bf16   v8bf;
typedef __attribute__((ext_vector_type(8)))  float    v8f;
typedef __attribute__((ext_vector_type(4)))  unsigned v4u;
typedef __attribute__((ext_vector_type(4)))  unsigned tdm_v4u;
typedef __attribute__((ext_vector_type(8)))  int      tdm_v8i;
typedef __attribute__((ext_vector_type(4)))  int      tdm_v4i;

#if defined(__has_builtin)
#  if __has_builtin(__builtin_amdgcn_tensor_load_to_lds) && \
      __has_builtin(__builtin_amdgcn_s_wait_tensorcnt)
#    define HAVE_TDM 1
#  endif
#endif
#ifndef HAVE_TDM
#  define HAVE_TDM 0
#endif

// float -> bf16 round-to-nearest-even
__device__ __forceinline__ __bf16 f2bf(float f) {
    union { float f; unsigned u; } v; v.f = f;
    unsigned r = v.u + 0x7FFFu + ((v.u >> 16) & 1u);
    unsigned short h = (unsigned short)(r >> 16);
    return __builtin_bit_cast(__bf16, h);
}

// padded sA element offset for logical element ee (0..767) of row `row`
__device__ __forceinline__ int aoff(int row, int ee) {
    return row * AROW_ + ee + ((ee >> 8) << 3);   // 8-elem pad per 256 elems
}

// Assemble a 16-element bf16 WMMA fragment from two 16-byte runs.
__device__ __forceinline__ v16bf frag16(const __bf16* p0, const __bf16* p1) {
    v8bf lo = __builtin_bit_cast(v8bf, *(const v4u*)p0);
    v8bf hi = __builtin_bit_cast(v8bf, *(const v4u*)p1);
    v16bf r;
#pragma unroll
    for (int i = 0; i < 8; ++i) { r[i] = lo[i]; r[i + 8] = hi[i]; }
    return r;
}

__device__ __forceinline__ v8f zero8() {
    v8f a;
#pragma unroll
    for (int i = 0; i < 8; ++i) a[i] = 0.0f;
    return a;
}

// ---------------------------------------------------------------------------
// 0) tiny accumulator init
// ---------------------------------------------------------------------------
__global__ void init_kernel(int* counts, float* gacc, float* scal) {
    int i = threadIdx.x;
    if (i < E_) { counts[i] = 0; gacc[i] = 0.0f; }
    if (i < 2)  scal[i] = 0.0f;
}

// ---------------------------------------------------------------------------
// 1) gating: logits = leaky_relu(x @ gate_w + gate_b); top-2 + softmax;
//    build per-expert token lists; accumulate stats for zloss / cvloss.
// ---------------------------------------------------------------------------
__global__ __launch_bounds__(256) void gate_kernel(
    const float* __restrict__ x, const float* __restrict__ gw,
    const float* __restrict__ gb,
    int* __restrict__ counts, int* __restrict__ tok, float* __restrict__ prb,
    float* __restrict__ gacc, float* __restrict__ scal) {
    int t = blockIdx.x * 256 + threadIdx.x;
    if (t >= N_) return;
    const float* xr = x + (size_t)t * D_;
    float acc[E_];
#pragma unroll
    for (int e = 0; e < E_; ++e) acc[e] = gb[e];
    for (int d = 0; d < D_; ++d) {
        float xv = xr[d];
#pragma unroll
        for (int e = 0; e < E_; ++e) acc[e] += xv * gw[d * E_ + e];
    }
    float zs = 0.0f;
    float v0 = -3.4e38f, v1 = -3.4e38f; int i0 = 0, i1 = 0;
#pragma unroll
    for (int e = 0; e < E_; ++e) {
        float v = acc[e];
        v = (v >= 0.0f) ? v : 0.01f * v;     // leaky_relu
        zs += v * v;
        if (v > v0)      { v1 = v0; i1 = i0; v0 = v; i0 = e; }
        else if (v > v1) { v1 = v; i1 = e; }
    }
    float e1 = __expf(v1 - v0);
    float p0 = 1.0f / (1.0f + e1);
    float p1 = e1 * p0;
    atomicAdd(&scal[0], zs);
    atomicAdd(&scal[1], p0 * p0 + p1 * p1);
    atomicAdd(&gacc[i0], p0);
    atomicAdd(&gacc[i1], p1);
    int s0 = atomicAdd(&counts[i0], 1);
    tok[i0 * N_ + s0] = t; prb[i0 * N_ + s0] = p0;
    int s1 = atomicAdd(&counts[i1], 1);
    tok[i1 * N_ + s1] = t; prb[i1 * N_ + s1] = p1;
}

// ---------------------------------------------------------------------------
// 2) LayerNorm -> bf16 activations
// ---------------------------------------------------------------------------
__global__ __launch_bounds__(256) void ln_kernel(
    const float* __restrict__ x, const float* __restrict__ lnw,
    const float* __restrict__ lnb, __bf16* __restrict__ xn) {
    int t = blockIdx.x;
    int tid = threadIdx.x;
    const float* xr = x + (size_t)t * D_;
    float s = 0.0f, s2 = 0.0f;
    for (int i = tid; i < D_; i += 256) { float v = xr[i]; s += v; s2 += v * v; }
    __shared__ float rs[256], rq[256];
    rs[tid] = s; rq[tid] = s2;
    __syncthreads();
    for (int off = 128; off > 0; off >>= 1) {
        if (tid < off) { rs[tid] += rs[tid + off]; rq[tid] += rq[tid + off]; }
        __syncthreads();
    }
    float mean = rs[0] * (1.0f / D_);
    float var  = rq[0] * (1.0f / D_) - mean * mean;
    float inv  = rsqrtf(var + 1e-5f);
    for (int i = tid; i < D_; i += 256) {
        float v = (xr[i] - mean) * inv * lnw[i] + lnb[i];
        xn[(size_t)t * D_ + i] = f2bf(v);
    }
}

// ---------------------------------------------------------------------------
// 3) weight conversion f32 -> bf16, transposed for contiguous WMMA B frags.
//    w1 [E][D][H] -> w1t [E][H][D] ; w2 [E][H][D] -> w2t [E][D][H]
// ---------------------------------------------------------------------------
__global__ __launch_bounds__(256) void convert_weights(
    const float* __restrict__ w1, const float* __restrict__ w2,
    __bf16* __restrict__ w1t, __bf16* __restrict__ w2t) {
    size_t i = (size_t)blockIdx.x * 256 + threadIdx.x;
    if (i >= (size_t)E_ * D_ * H_) return;
    {   // w1
        size_t e = i / ((size_t)D_ * H_);
        size_t r = i % ((size_t)D_ * H_);
        size_t d = r / H_, h = r % H_;
        w1t[e * (size_t)H_ * D_ + h * D_ + d] = f2bf(w1[i]);
    }
    {   // w2 (same flat range: E*H*D)
        size_t e = i / ((size_t)H_ * D_);
        size_t r = i % ((size_t)H_ * D_);
        size_t h = r / D_, d = r % D_;
        w2t[e * (size_t)D_ * H_ + d * H_ + h] = f2bf(w2[i]);
    }
}

// ---------------------------------------------------------------------------
// 4) expert FFN tiles: y += prob * (relu(xn @ W1 + b1) @ W2 + b2)
//    32-token tiles (two 16-row WMMA sub-tiles sharing each B fragment),
//    A-tile gathered into LDS by the Tensor Data Mover (gather mode, with
//    LDS padding), bf16 WMMA 16x16x32, 8 waves/block, padded h-tile in LDS.
// ---------------------------------------------------------------------------
__global__ __launch_bounds__(256) void moe_ffn_kernel(
    const __bf16* __restrict__ xn,
    const __bf16* __restrict__ w1t, const __bf16* __restrict__ w2t,
    const float* __restrict__ b1, const float* __restrict__ b2,
    const int* __restrict__ counts, const int* __restrict__ tok,
    const float* __restrict__ prb, float* __restrict__ y) {

    const int e    = blockIdx.x / TILES_;
    const int tile = blockIdx.x % TILES_;
    const int cnt  = counts[e];
    const int row0 = tile * MT_;
    if (row0 >= cnt) return;

    __shared__ __align__(16) __bf16 sA[MT_ * AROW_];   // ~49.5 KB padded A tile
    __shared__ __align__(16) __bf16 sH[MT_ * HROW_];   // ~96.5 KB padded h tile
    __shared__ int   sTok[MT_];
    __shared__ float sPrb[MT_];

    const int tid  = threadIdx.x;
    const int wave = tid >> 5;
    const int lane = tid & 31;
    const int half = lane >> 4;    // 0: lanes 0-15, 1: lanes 16-31
    const int l16  = lane & 15;
    const int kbA  = half * 8;     // A-frag per-lane K base within 32-block
    const int kbB  = half * 8;     // B-frag per-lane K base within 32-block

    if (tid < MT_) {
        int idx = row0 + tid;
        bool valid = idx < cnt;
        sTok[tid] = tok[e * N_ + (valid ? idx : row0)];
        sPrb[tid] = valid ? prb[e * N_ + idx] : 0.0f;
    }
    __syncthreads();

    // ---- stage A-tile (32 gathered token rows of xn) into LDS ----
#if HAVE_TDM
    if (wave == 0) {
        const unsigned long long ga = (unsigned long long)(uintptr_t)xn;
        const unsigned ldsA = (unsigned)(uintptr_t)(void*)sA;
        for (int hb = 0; hb < 2; ++hb) {          // rows 0-15, rows 16-31
            tdm_v4u g0;
            g0[0] = 1u | (1u << 31);              // count=1, gather_mode, 16b idx
            g0[1] = ldsA + (unsigned)(hb * 16 * AROW_ * 2);
            g0[2] = (unsigned)(ga & 0xFFFFFFFFu);
            g0[3] = (unsigned)((ga >> 32) & 0x1FFFFFFu) | (2u << 30); // type=2
            tdm_v8i g1;
            // data_size=2B | pad_enable | pad_interval=6 (128 dw) |
            // pad_amount=3 (4 dw) -> +8 bf16 elems per 256 elems
            g1[0] = (int)((1u << 16) | (1u << 20) | (6u << 22) | (3u << 25));
            g1[1] = (int)((unsigned)D_ << 16);    // tensor_dim0 low16
            g1[2] = (int)((unsigned)N_ << 16);    // dim0 hi=0 | tensor_dim1 low16
            g1[3] = (int)((unsigned)D_ << 16);    // dim1 hi=0 | tile_dim0
            g1[4] = 16;                           // tile_dim1 = #valid indices
            g1[5] = (int)D_;                      // tensor_dim0_stride low32
            g1[6] = 0;
            g1[7] = 0;
            tdm_v4i g2, g3;
#pragma unroll
            for (int j = 0; j < 4; ++j) {
                int r  = hb * 16 + j * 2;
                int r2 = r + 8;
                g2[j] = (sTok[r]  & 0xFFFF) | (sTok[r  + 1] << 16);
                g3[j] = (sTok[r2] & 0xFFFF) | (sTok[r2 + 1] << 16);
            }
#if __clang_major__ >= 23
            tdm_v8i g4;
#pragma unroll
            for (int j = 0; j < 8; ++j) g4[j] = 0;
            __builtin_amdgcn_tensor_load_to_lds(g0, g1, g2, g3, g4, 0);
#else
            __builtin_amdgcn_tensor_load_to_lds(g0, g1, g2, g3, 0);
#endif
        }
        __builtin_amdgcn_s_wait_tensorcnt(0);
    }
#else
    for (int i = tid; i < MT_ * (D_ / 8); i += 256) {
        int r = i / (D_ / 8), c = (i % (D_ / 8)) * 8;
        *(v4u*)&sA[aoff(r, c)] = *(const v4u*)&xn[(size_t)sTok[r] * D_ + c];
    }
#endif
    __syncthreads();

    // ---- phase 1: h = relu(xn_tile @ W1[e] + b1[e]) -> LDS (bf16) ----
    const __bf16* w1e = w1t + (size_t)e * H_ * D_;
    for (int nb = wave; nb < H_ / 16; nb += 8) {
        const int col = nb * 16 + l16;
        const __bf16* bCol = w1e + (size_t)col * D_;
        __builtin_prefetch(bCol, 0, 1);
        v8f acc0 = zero8(), acc1 = zero8();
#pragma unroll 4
        for (int kk = 0; kk < D_ / 32; ++kk) {
            const int k0 = kk * 32;
            v16bf b  = frag16(bCol + k0 + kbB, bCol + k0 + 16 + kbB);
            v16bf a0 = frag16(&sA[aoff(l16, k0 + kbA)],
                              &sA[aoff(l16, k0 + 16 + kbA)]);
            v16bf a1 = frag16(&sA[aoff(16 + l16, k0 + kbA)],
                              &sA[aoff(16 + l16, k0 + 16 + kbA)]);
            acc0 = __builtin_amdgcn_wmma_f32_16x16x32_bf16(
                false, a0, false, b, (short)0, acc0, false, false);
            acc1 = __builtin_amdgcn_wmma_f32_16x16x32_bf16(
                false, a1, false, b, (short)0, acc1, false, false);
        }
        const float bias = b1[e * H_ + col];
#pragma unroll
        for (int v = 0; v < 8; ++v) {
            const int m = v + half * 8;
            float v0 = acc0[v] + bias; v0 = v0 > 0.0f ? v0 : 0.0f;
            float v1 = acc1[v] + bias; v1 = v1 > 0.0f ? v1 : 0.0f;
            sH[m * HROW_ + col]        = f2bf(v0);
            sH[(16 + m) * HROW_ + col] = f2bf(v1);
        }
    }
    __syncthreads();

    // ---- phase 2: o = h @ W2[e] + b2[e]; combine with prob via atomics ----
    const __bf16* w2e = w2t + (size_t)e * D_ * H_;
    for (int nb = wave; nb < D_ / 16; nb += 8) {
        const int col = nb * 16 + l16;
        const __bf16* bCol = w2e + (size_t)col * H_;
        __builtin_prefetch(bCol, 0, 1);
        v8f acc0 = zero8(), acc1 = zero8();
#pragma unroll 4
        for (int kk = 0; kk < H_ / 32; ++kk) {
            const int k0 = kk * 32;
            v16bf b  = frag16(bCol + k0 + kbB, bCol + k0 + 16 + kbB);
            v16bf a0 = frag16(&sH[l16 * HROW_ + k0 + kbA],
                              &sH[l16 * HROW_ + k0 + 16 + kbA]);
            v16bf a1 = frag16(&sH[(16 + l16) * HROW_ + k0 + kbA],
                              &sH[(16 + l16) * HROW_ + k0 + 16 + kbA]);
            acc0 = __builtin_amdgcn_wmma_f32_16x16x32_bf16(
                false, a0, false, b, (short)0, acc0, false, false);
            acc1 = __builtin_amdgcn_wmma_f32_16x16x32_bf16(
                false, a1, false, b, (short)0, acc1, false, false);
        }
        const float bias = b2[e * D_ + col];
#pragma unroll
        for (int v = 0; v < 8; ++v) {
            const int m = v + half * 8;
            if (row0 + m < cnt) {
                float val = (acc0[v] + bias) * sPrb[m];
                atomicAdd(&y[(size_t)sTok[m] * D_ + col], val);
            }
            if (row0 + 16 + m < cnt) {
                float val = (acc1[v] + bias) * sPrb[16 + m];
                atomicAdd(&y[(size_t)sTok[16 + m] * D_ + col], val);
            }
        }
    }
}

// ---------------------------------------------------------------------------
// 5) scalars: gates_sum, zloss, cvloss
// ---------------------------------------------------------------------------
__global__ void finalize_kernel(const float* gacc, const float* scal, float* out) {
    if (threadIdx.x == 0 && blockIdx.x == 0) {
        float total = 0.0f;
        for (int e = 0; e < E_; ++e) { out[e] = gacc[e]; total += gacc[e]; }
        const float NE = (float)N_ * (float)E_;
        out[E_] = scal[0] / NE;                       // zloss
        float mean = total / NE;
        float var  = (scal[1] - NE * mean * mean) / (NE - 1.0f);   // unbiased
        out[E_ + 1] = var / (mean * mean + 1e-10f);   // cvloss
    }
}

// ---------------------------------------------------------------------------
extern "C" void kernel_launch(void* const* d_in, const int* in_sizes, int n_in,
                              void* d_out, int out_size, void* d_ws, size_t ws_size,
                              hipStream_t stream) {
    const float* x   = (const float*)d_in[0];
    const float* gw  = (const float*)d_in[1];
    const float* gb  = (const float*)d_in[2];
    const float* lnw = (const float*)d_in[3];
    const float* lnb = (const float*)d_in[4];
    const float* w1  = (const float*)d_in[5];
    const float* b1  = (const float*)d_in[6];
    const float* w2  = (const float*)d_in[7];
    const float* b2  = (const float*)d_in[8];
    float* out = (float*)d_out;                 // [y | gates_sum | zloss | cvloss]

    char* ws = (char*)d_ws;
    size_t off = 0;
    auto carve = [&](size_t bytes) -> void* {
        void* p = ws + off;
        off = (off + bytes + 255) & ~(size_t)255;
        return p;
    };
    __bf16* xn     = (__bf16*)carve((size_t)N_ * D_ * 2);
    __bf16* w1t    = (__bf16*)carve((size_t)E_ * H_ * D_ * 2);
    __bf16* w2t    = (__bf16*)carve((size_t)E_ * D_ * H_ * 2);
    int*    counts = (int*)   carve((size_t)E_ * 4);
    int*    tokl   = (int*)   carve((size_t)E_ * N_ * 4);
    float*  prbl   = (float*) carve((size_t)E_ * N_ * 4);
    float*  gacc   = (float*) carve((size_t)E_ * 4);
    float*  scal   = (float*) carve(2 * 4);
    (void)in_sizes; (void)n_in; (void)out_size; (void)ws_size;

    // residual: y = x
    hipMemcpyAsync(out, x, (size_t)N_ * D_ * sizeof(float),
                   hipMemcpyDeviceToDevice, stream);

    init_kernel<<<1, 64, 0, stream>>>(counts, gacc, scal);
    gate_kernel<<<N_ / 256, 256, 0, stream>>>(x, gw, gb, counts, tokl, prbl,
                                              gacc, scal);
    ln_kernel<<<N_, 256, 0, stream>>>(x, lnw, lnb, xn);
    {
        size_t total = (size_t)E_ * D_ * H_;
        int blocks = (int)((total + 255) / 256);
        convert_weights<<<blocks, 256, 0, stream>>>(w1, w2, w1t, w2t);
    }
    moe_ffn_kernel<<<E_ * TILES_, 256, 0, stream>>>(xn, w1t, w2t, b1, b2,
                                                    counts, tokl, prbl, out);
    finalize_kernel<<<1, 32, 0, stream>>>(gacc, scal, out + (size_t)N_ * D_);
}
